// LSTM_71081708749284
// MI455X (gfx1250) — matrix operand order repaired and checked by reading, not compiled, not run on hardware
//
#include <hip/hip_runtime.h>
#include <stdint.h>
#include <stddef.h>

// Problem dims (match reference)
#define B_  64
#define S_  512
#define V_  32000
#define E_  512
#define H_  1024
#define G_  (4 * H_)   // 4096

// ---------------------------------------------------------------------------
// WMMA types (CDNA5 / gfx1250, wave32)
// ---------------------------------------------------------------------------
typedef __attribute__((ext_vector_type(16))) __bf16 v16bf;
typedef __attribute__((ext_vector_type(8)))  float  v8f;
typedef __attribute__((ext_vector_type(4)))  uint32_t u32x4;
typedef __attribute__((ext_vector_type(8)))  int32_t  i32x8;
typedef __attribute__((ext_vector_type(4)))  int32_t  i32x4;

union Frag16 { v16bf v; uint32_t u[8]; };

__device__ __forceinline__ uint16_t f32_to_bf16_rne(float f) {
  union { float f; uint32_t u; } x; x.f = f;
  uint32_t r = x.u + 0x7FFFu + ((x.u >> 16) & 1u);
  return (uint16_t)(r >> 16);
}

__device__ __forceinline__ float sigmoidf_fast(float x) {
  return 1.0f / (1.0f + __expf(-x));
}

// The TDM writes LDS behind the compiler's back (descriptor carries only a
// byte offset). Escape the tile pointer through an asm with a memory clobber
// so the compiler must treat the tile as written and keep the ds_loads.
__device__ __forceinline__ void lds_publish(const void* p) {
  asm volatile("" :: "v"(p) : "memory");
}

// A fragment (16x32 bf16), row-major source. rowPtr = base of this lane's row
// (row m = lane&15). Layout: lanes 0-15 hold K 0..15 (v0..v7); lanes 16-31
// hold K 16..31, two bf16 packed per VGPR (matches B fragment k order).
__device__ __forceinline__ v16bf load_a_frag(const uint16_t* rowPtr, int kb, int lane) {
  const int hi = lane >> 4;
  Frag16 r;
#pragma unroll
  for (int v = 0; v < 8; ++v) {
    const int k = (hi << 4) + (v << 1);
    r.u[v] = *reinterpret_cast<const uint32_t*>(rowPtr + kb + k);
  }
  return r.v;
}

// B fragment (32x16 bf16) from a K-major (transposed) weight matrix in global:
// srcT is [N][K] row-major, ldk = K.
__device__ __forceinline__ v16bf load_b_frag(const uint16_t* srcT, int ldk,
                                             int n0, int kb, int lane) {
  const int n  = lane & 15;
  const int hi = lane >> 4;
  const uint16_t* p = srcT + (size_t)(n0 + n) * ldk + kb + (hi << 4);
  Frag16 r;
#pragma unroll
  for (int v = 0; v < 8; ++v) {
    r.u[v] = *reinterpret_cast<const uint32_t*>(p + (v << 1));
  }
  return r.v;
}

// B fragment from an LDS-staged K-major tile (row length = ldk elements).
__device__ __forceinline__ v16bf load_b_frag_lds(const uint16_t* tile, int ldk,
                                                 int n0, int kb, int lane) {
  const int n  = lane & 15;
  const int hi = lane >> 4;
  const uint16_t* p = tile + (n0 + n) * ldk + kb + (hi << 4);
  Frag16 r;
#pragma unroll
  for (int v = 0; v < 8; ++v) {
    r.u[v] = *reinterpret_cast<const uint32_t*>(p + (v << 1));
  }
  return r.v;
}

__device__ __forceinline__ v8f wmma_bf16(v16bf a, v16bf b, v8f c) {
  // (neg_a, A, neg_b, B, c_mod, C, reuse_a, reuse_b)
  return __builtin_amdgcn_wmma_f32_16x16x32_bf16(false, a, false, b,
                                                 (short)0, c, false, false);
}

// ---------------------------------------------------------------------------
// Tensor Data Mover: 2D tile load (global, K-major bf16) -> LDS.
// D# per CDNA5 ISA:
//   group0: count=1 | lds_addr | global_addr | type=2
//   group1: data_size=1(2B) | tensor_dim0/1 | tile_dim0/1 | dim0_stride
// dims/strides in data_size units. 2D tensor => groups 2/3 zero (VADDR2/3).
// ---------------------------------------------------------------------------
__device__ __forceinline__ void tdm_load_2d_bf16(uint32_t lds_byte_off,
                                                 const uint16_t* gptr,
                                                 uint32_t tensor_d0, uint32_t tensor_d1,
                                                 uint32_t tile_d0, uint32_t tile_d1,
                                                 uint32_t d0_stride) {
  const uint64_t ga = (uint64_t)(uintptr_t)gptr;
  u32x4 g0;
  g0.x = 1u;                                           // count=1, user descriptor
  g0.y = lds_byte_off;                                 // LDS byte address
  g0.z = (uint32_t)ga;                                 // global_addr[31:0]
  g0.w = (uint32_t)((ga >> 32) & 0x01FFFFFFu) | 0x80000000u; // ga[56:32] | type=2

  i32x8 g1;
  g1[0] = (int32_t)(1u << 16);                         // wg_mask=0, data_size=1 (2B)
  g1[1] = (int32_t)((tensor_d0 & 0xFFFFu) << 16);      // tensor_dim0[15:0] @ [63:48]
  g1[2] = (int32_t)(((tensor_d0 >> 16) & 0xFFFFu) |
                    ((tensor_d1 & 0xFFFFu) << 16));    // dim0 hi | dim1 lo
  g1[3] = (int32_t)(((tensor_d1 >> 16) & 0xFFFFu) |
                    ((tile_d0 & 0xFFFFu) << 16));      // dim1 hi | tile_dim0
  g1[4] = (int32_t)(tile_d1 & 0xFFFFu);                // tile_dim1 | tile_dim2=0
  g1[5] = (int32_t)d0_stride;                          // tensor_dim0_stride[31:0]
  g1[6] = 0;                                           // stride hi | dim1_stride lo
  g1[7] = 0;                                           // dim1_stride hi

  const i32x4 gz = {0, 0, 0, 0};
#if __clang_major__ >= 23
  const i32x8 gz8 = {0, 0, 0, 0, 0, 0, 0, 0};
  __builtin_amdgcn_tensor_load_to_lds(g0, g1, gz, gz, gz8, 0);
#else
  __builtin_amdgcn_tensor_load_to_lds(g0, g1, gz, gz, 0);
#endif
}

// ---------------------------------------------------------------------------
// Weight conversion kernels (fp32 -> bf16, optionally transposed to K-major)
// ---------------------------------------------------------------------------
__global__ void k_convert_bf16(const float* __restrict__ src,
                               uint16_t* __restrict__ dst, size_t n) {
  for (size_t i = (size_t)blockIdx.x * blockDim.x + threadIdx.x; i < n;
       i += (size_t)gridDim.x * blockDim.x) {
    dst[i] = f32_to_bf16_rne(src[i]);
  }
}

// src [R][C] fp32 row-major -> dst [C][R] bf16 row-major (i.e. K-major weights)
__global__ void k_transpose_bf16(const float* __restrict__ src,
                                 uint16_t* __restrict__ dst, int R, int C) {
  const size_t n = (size_t)R * C;
  for (size_t i = (size_t)blockIdx.x * blockDim.x + threadIdx.x; i < n;
       i += (size_t)gridDim.x * blockDim.x) {
    const size_t r = i / C, c = i % C;
    dst[c * (size_t)R + r] = f32_to_bf16_rne(src[i]);
  }
}

__global__ void k_init_state(float* __restrict__ h, float* __restrict__ c,
                             uint16_t* __restrict__ hB) {
  const int idx = blockIdx.x * blockDim.x + threadIdx.x;
  if (idx < B_ * H_) { h[idx] = 0.0f; c[idx] = 0.0f; hB[idx] = 0; }
}

// ---------------------------------------------------------------------------
// Phase 1: xp[s*B+b][g] = emb[x[b][s]] @ Wx + bx + bh   (M=B*S, N=4H, K=E)
// ---------------------------------------------------------------------------
__global__ void k_xp_gemm(const int* __restrict__ x,
                          const uint16_t* __restrict__ embB,
                          const uint16_t* __restrict__ WxT,   // [4H][E] bf16
                          const float* __restrict__ bx,
                          const float* __restrict__ bh,
                          float* __restrict__ xp) {           // [S*B][4H]
  const int NU   = G_ / 64;  // 64 wave-strips along N
  const int lane = threadIdx.x & 31;
  const int wid  = blockIdx.x * (blockDim.x >> 5) + (threadIdx.x >> 5);
  const int mt = wid / NU, nu = wid % NU;
  const int m0 = mt * 16, n0base = nu * 64;

  // Embedding gather: this lane's A row (row m = lane&15)
  const int row = m0 + (lane & 15);
  const int s = row / B_, b = row % B_;
  const int tok = x[b * S_ + s];
  const uint16_t* arow = embB + (size_t)tok * E_;

  v8f acc[4] = {};
  for (int kb = 0; kb < E_; kb += 32) {
    __builtin_prefetch(arow + kb + 32, 0, 1);
    const v16bf a = load_a_frag(arow, kb, lane);
#pragma unroll
    for (int i = 0; i < 4; ++i) {
      const v16bf bf = load_b_frag(WxT, E_, n0base + 16 * i, kb, lane);
      acc[i] = wmma_bf16(a, bf, acc[i]);
    }
  }

  const int n = lane & 15, hi = lane >> 4;
#pragma unroll
  for (int i = 0; i < 4; ++i) {
#pragma unroll
    for (int r = 0; r < 8; ++r) {
      const int mm = m0 + r + (hi << 3);
      const int nn = n0base + 16 * i + n;
      xp[(size_t)mm * G_ + nn] = acc[i][r] + bx[nn] + bh[nn];
    }
  }
}

// ---------------------------------------------------------------------------
// Phase 2a (per step): gates = xp_t + h @ Wh   (M=64, N=4H, K=H)
// TDM-staged: each block owns a 64x128 output tile; the Tensor Data Mover
// double-buffers 128x64 Wh slabs (16 KB) into LDS, shared by all 8 waves.
// ---------------------------------------------------------------------------
#define KC_   64                    // K elements per TDM chunk
#define NBLK_ 128                   // N columns per block
#define NCHUNK_ (H_ / KC_)          // 16 chunks

__global__ void k_gates_gemm(const uint16_t* __restrict__ hB,    // [B][H] bf16
                             const uint16_t* __restrict__ WhT,   // [4H][H] bf16
                             const float* __restrict__ xpt,      // [B][4H]
                             float* __restrict__ gates) {        // [B][4H]
  __shared__ uint16_t ldsB[2][NBLK_ * KC_];          // 2 x 16 KB

  const int lane = threadIdx.x & 31;
  const int wave = threadIdx.x >> 5;                 // 0..7
  const int mt    = wave >> 1;                       // 0..3  (M tile)
  const int strip = wave & 1;                        // 0..1  (64-col strip)
  const int m0     = mt * 16;
  const int nblk0  = blockIdx.x * NBLK_;             // block's first column
  const int n0base = strip * 64;                     // within LDS tile

  const uint16_t* arow = hB + (size_t)(m0 + (lane & 15)) * H_;
  const uint16_t* wblk = WhT + (size_t)nblk0 * H_;   // 128 rows x 1024 K slab

  // Deterministic LDS contents + visible writes (TDM writes bypass the
  // compiler's view of ldsB; without stores, the loads would fold to undef).
  {
    uint32_t* w = reinterpret_cast<uint32_t*>(&ldsB[0][0]);
    for (int i = threadIdx.x; i < NBLK_ * KC_; i += blockDim.x) w[i] = 0u;
  }
  __syncthreads();

  // Prime the pipeline: chunk 0 -> buffer 0 (one TDM issue per workgroup).
  if (wave == 0) {
    tdm_load_2d_bf16((uint32_t)0, wblk, H_, G_, KC_, NBLK_, H_);
  }

  v8f acc[4] = {};
  for (int ch = 0; ch < NCHUNK_; ++ch) {
    const int buf = ch & 1;
    if (wave == 0) {
      if (ch + 1 < NCHUNK_) {
        // Kick next chunk into the other buffer while we compute this one.
        tdm_load_2d_bf16((uint32_t)((ch + 1) & 1) * (NBLK_ * KC_ * 2),
                         wblk + (ch + 1) * KC_, H_, G_, KC_, NBLK_, H_);
        __builtin_amdgcn_s_wait_tensorcnt(1);   // oldest (chunk ch) landed
      } else {
        __builtin_amdgcn_s_wait_tensorcnt(0);   // drain on final chunk
      }
    }
    __syncthreads();                            // publish LDS tile to all waves
    lds_publish(&ldsB[buf][0]);                 // TDM-written: force real loads

    const int kb = ch * KC_;
#pragma unroll
    for (int ks = 0; ks < KC_; ks += 32) {
      const v16bf a = load_a_frag(arow, kb + ks, lane);
#pragma unroll
      for (int i = 0; i < 4; ++i) {
        const v16bf bf = load_b_frag_lds(ldsB[buf], KC_, n0base + 16 * i, ks, lane);
        acc[i] = wmma_bf16(a, bf, acc[i]);
      }
    }
    __syncthreads();                            // done reading before overwrite
  }

  const int n = lane & 15, hi = lane >> 4;
#pragma unroll
  for (int i = 0; i < 4; ++i) {
#pragma unroll
    for (int r = 0; r < 8; ++r) {
      const int mm = m0 + r + (hi << 3);
      const int nn = nblk0 + n0base + 16 * i + n;
      gates[(size_t)mm * G_ + nn] = acc[i][r] + xpt[(size_t)mm * G_ + nn];
    }
  }
}

// ---------------------------------------------------------------------------
// Phase 2b (per step): gate nonlinearities, update c/h, emit bf16 h.
// Chunk order f, i, g, o (matches jnp.split along 4H).
// ---------------------------------------------------------------------------
__global__ void k_pointwise(const float* __restrict__ gates,
                            float* __restrict__ h, float* __restrict__ c,
                            uint16_t* __restrict__ hB) {
  const int idx = blockIdx.x * blockDim.x + threadIdx.x;
  if (idx >= B_ * H_) return;
  const int b = idx >> 10;          // H_ == 1024
  const int j = idx & (H_ - 1);
  const float* g = gates + (size_t)b * G_;
  const float vf = sigmoidf_fast(g[j]);
  const float vi = sigmoidf_fast(g[H_ + j]);
  const float vg = tanhf(g[2 * H_ + j]);
  const float vo = sigmoidf_fast(g[3 * H_ + j]);
  const float cn = vf * c[idx] + vi * vg;
  const float hn = vo * tanhf(cn);
  c[idx] = cn;
  h[idx] = hn;
  hB[idx] = f32_to_bf16_rne(hn);
}

// ---------------------------------------------------------------------------
// Phase 3: logits = h @ Wout + bout   (M=64, N=32000, K=H)
// ---------------------------------------------------------------------------
__global__ void k_logits_gemm(const uint16_t* __restrict__ hB,     // [B][H]
                              const uint16_t* __restrict__ WoutT,  // [V][H]
                              const float* __restrict__ bout,
                              float* __restrict__ out) {           // [B][V]
  const int NU   = V_ / 64;  // 500
  const int lane = threadIdx.x & 31;
  const int wid  = blockIdx.x * (blockDim.x >> 5) + (threadIdx.x >> 5);
  const int mt = wid / NU, nu = wid % NU;
  const int m0 = mt * 16, n0base = nu * 64;

  const uint16_t* arow = hB + (size_t)(m0 + (lane & 15)) * H_;

  v8f acc[4] = {};
  for (int kb = 0; kb < H_; kb += 32) {
    const v16bf a = load_a_frag(arow, kb, lane);
#pragma unroll
    for (int i = 0; i < 4; ++i) {
      const v16bf bf = load_b_frag(WoutT, H_, n0base + 16 * i, kb, lane);
      acc[i] = wmma_bf16(a, bf, acc[i]);
    }
  }

  const int n = lane & 15, hi = lane >> 4;
#pragma unroll
  for (int i = 0; i < 4; ++i) {
#pragma unroll
    for (int r = 0; r < 8; ++r) {
      const int mm = m0 + r + (hi << 3);
      const int nn = n0base + 16 * i + n;
      out[(size_t)mm * V_ + nn] = acc[i][r] + bout[nn];
    }
  }
}

__global__ void k_copy_hc(const float* __restrict__ h,
                          const float* __restrict__ c,
                          float* __restrict__ out) {
  const int idx = blockIdx.x * blockDim.x + threadIdx.x;
  if (idx >= B_ * H_) return;
  out[(size_t)B_ * V_ + idx] = h[idx];
  out[(size_t)B_ * V_ + (size_t)B_ * H_ + idx] = c[idx];
}

// ---------------------------------------------------------------------------
// Host driver
// ---------------------------------------------------------------------------
extern "C" void kernel_launch(void* const* d_in, const int* in_sizes, int n_in,
                              void* d_out, int out_size, void* d_ws, size_t ws_size,
                              hipStream_t stream) {
  (void)in_sizes; (void)n_in; (void)out_size; (void)ws_size;

  const int*   x    = (const int*)  d_in[0];
  const float* emb  = (const float*)d_in[1];
  const float* Wx   = (const float*)d_in[2];
  const float* bx   = (const float*)d_in[3];
  const float* Wh   = (const float*)d_in[4];
  const float* bh   = (const float*)d_in[5];
  const float* Wout = (const float*)d_in[6];
  const float* bout = (const float*)d_in[7];
  float* out = (float*)d_out;

  // Workspace carve-out (256B aligned)
  char* ws = (char*)d_ws;
  size_t off = 0;
  auto carve = [&](size_t bytes) -> void* {
    void* p = ws + off;
    off += (bytes + 255) & ~(size_t)255;
    return p;
  };
  uint16_t* embB  = (uint16_t*)carve((size_t)V_ * E_ * 2);   // emb  bf16 [V][E]
  uint16_t* WxT   = (uint16_t*)carve((size_t)G_ * E_ * 2);   // Wx^T bf16 [4H][E]
  uint16_t* WhT   = (uint16_t*)carve((size_t)G_ * H_ * 2);   // Wh^T bf16 [4H][H]
  uint16_t* WoutT = (uint16_t*)carve((size_t)V_ * H_ * 2);   // Wout^T   [V][H]
  float*    xp    = (float*)   carve((size_t)S_ * B_ * G_ * 4);
  float*    gates = (float*)   carve((size_t)B_ * G_ * 4);
  float*    h     = (float*)   carve((size_t)B_ * H_ * 4);
  float*    c     = (float*)   carve((size_t)B_ * H_ * 4);
  uint16_t* hB    = (uint16_t*)carve((size_t)B_ * H_ * 2);

  const int T = 256;

  // Weight / embedding conversion (bf16, weights K-major for WMMA B frags)
  k_convert_bf16  <<<2048, T, 0, stream>>>(emb, embB, (size_t)V_ * E_);
  k_transpose_bf16<<<2048, T, 0, stream>>>(Wx,   WxT,   E_, G_);
  k_transpose_bf16<<<2048, T, 0, stream>>>(Wh,   WhT,   H_, G_);
  k_transpose_bf16<<<4096, T, 0, stream>>>(Wout, WoutT, H_, V_);
  k_init_state<<<(B_ * H_ + T - 1) / T, T, 0, stream>>>(h, c, hB);

  // Phase 1: time-parallel input projection (M=32768 -> 2048 tiles, NU=64)
  k_xp_gemm<<<(2048 * 64) / 8, T, 0, stream>>>(x, embB, WxT, bx, bh, xp);

  // Phase 2: sequential scan; Wh slabs staged through LDS by the TDM.
  for (int t = 0; t < S_; ++t) {
    k_gates_gemm<<<G_ / NBLK_, T, 0, stream>>>(hB, WhT,
                                               xp + (size_t)t * B_ * G_, gates);
    k_pointwise<<<(B_ * H_ + T - 1) / T, T, 0, stream>>>(gates, h, c, hB);
  }

  // Phase 3: output projection + emit h, c
  k_logits_gemm<<<(4 * 500) / 8, T, 0, stream>>>(hB, WoutT, bout, out);
  k_copy_hc<<<(B_ * H_ + T - 1) / T, T, 0, stream>>>(h, c, out);
}
